// SpatioTemporalAttention_38955353375033
// MI455X (gfx1250) — matrix operand order repaired and checked by reading, not compiled, not run on hardware
//
#include <hip/hip_runtime.h>
#include <hip/hip_bf16.h>
#include <cstdint>
#include <cstddef>

// ---------------------------------------------------------------------------
// Types for CDNA5 WMMA (wave32, 16x16x32 f16 -> f32)
// ---------------------------------------------------------------------------
typedef __attribute__((ext_vector_type(16))) _Float16 v16h;
typedef __attribute__((ext_vector_type(8)))  _Float16 v8h;
typedef __attribute__((ext_vector_type(8)))  float    v8f;

__device__ __forceinline__ v8f wmma_f16(v16h a, v16h b, v8f c) {
    // D = A(16x32) * B(32x16) + C(16x16), f32 accumulate
    return __builtin_amdgcn_wmma_f32_16x16x32_f16(
        /*neg_a=*/false, a, /*neg_b=*/false, b,
        /*c_mod=*/(short)0, c, /*reuse_a=*/false, /*reuse_b=*/false);
}

// A fragment (16x32, MxK), base = &A[row0*ld + k0], row-major A.
// ISA layout: lane holds row m=lane&15; halves j=0..7 -> K=8*hi+j,
//             j=8..15 -> K=16+8*hi+(j-8). Two contiguous 16B chunks.
__device__ __forceinline__ v16h load_a_frag(const _Float16* base, int ld, int lane) {
    int m = lane & 15, hi = lane >> 4;
    const _Float16* p = base + (size_t)m * ld + 8 * hi;
    v8h lo = *(const v8h*)p;
    v8h hv = *(const v8h*)(p + 16);
    v16h r;
#pragma unroll
    for (int i = 0; i < 8; ++i) { r[i] = lo[i]; r[8 + i] = hv[i]; }
    return r;
}

// B fragment (32x16, KxN) taken from a ROW-MAJOR [N,K] matrix:
// B[k][n] = W[n0+n][k0+k]. base = &W[n0*ld + k0].
// ISA layout: lane holds col n=lane&15; halves j=0..7 -> K=8*hi+j,
//             j=8..15 -> K=16+8*hi+(j-8). Two contiguous 16B chunks.
__device__ __forceinline__ v16h load_b_frag(const _Float16* base, int ld, int lane) {
    int n = lane & 15, hi = lane >> 4;
    const _Float16* p = base + (size_t)n * ld + 8 * hi;
    v8h lo = *(const v8h*)p;
    v8h hv = *(const v8h*)(p + 16);
    v16h r;
#pragma unroll
    for (int i = 0; i < 8; ++i) { r[i] = lo[i]; r[8 + i] = hv[i]; }
    return r;
}

// ---------------------------------------------------------------------------
// Model constants
// ---------------------------------------------------------------------------
#define D_MODEL   1536
#define QKV_OUT   2304
#define NUM_HEADS 16
#define QG        4
#define HEAD_DIM  96
#define T_GRID    32
#define S_PAT     256
#define BATCH     2
#define M_ROWS    (BATCH * T_GRID * S_PAT)   // 16384
#define N_SEQ     (T_GRID * S_PAT)           // 8192

// ---------------------------------------------------------------------------
// f32 -> f16 convert
// ---------------------------------------------------------------------------
__global__ void cvt_f16_kernel(const float* __restrict__ src,
                               _Float16* __restrict__ dst, size_t n) {
    size_t i = (size_t)blockIdx.x * blockDim.x + threadIdx.x;
    if (i < n) dst[i] = (_Float16)src[i];
}

// ---------------------------------------------------------------------------
// GEMM: C[M,N] = A[M,K] * W[N,K]^T   (A,W f16 row-major; C f16 or f32)
// Block: 256 thr = 8 waves, 2(M) x 4(N) wave grid; wave tile 64x64 (4x4 WMMA).
// ---------------------------------------------------------------------------
template <bool HALF_OUT>
__global__ __launch_bounds__(256)
void gemm_kernel(const _Float16* __restrict__ A, const _Float16* __restrict__ W,
                 void* __restrict__ Cout, int M, int N, int K) {
    int lane = threadIdx.x & 31;
    int wave = threadIdx.x >> 5;
    int wm = wave & 1, wn = wave >> 1;
    int row0 = blockIdx.y * 128 + wm * 64;
    int col0 = blockIdx.x * 256 + wn * 64;

    v8f c[4][4] = {};
    for (int k0 = 0; k0 < K; k0 += 32) {
        v16h a[4], b[4];
#pragma unroll
        for (int mi = 0; mi < 4; ++mi)
            a[mi] = load_a_frag(A + (size_t)(row0 + mi * 16) * K + k0, K, lane);
#pragma unroll
        for (int nj = 0; nj < 4; ++nj)
            b[nj] = load_b_frag(W + (size_t)(col0 + nj * 16) * K + k0, K, lane);
#pragma unroll
        for (int mi = 0; mi < 4; ++mi)
#pragma unroll
            for (int nj = 0; nj < 4; ++nj)
                c[mi][nj] = wmma_f16(a[mi], b[nj], c[mi][nj]);
    }

    int n = lane & 15, hi = lane >> 4;
#pragma unroll
    for (int mi = 0; mi < 4; ++mi)
#pragma unroll
        for (int nj = 0; nj < 4; ++nj)
#pragma unroll
            for (int r = 0; r < 8; ++r) {
                size_t row = row0 + mi * 16 + r + 8 * hi;
                size_t col = col0 + nj * 16 + n;
                if (HALF_OUT)
                    ((_Float16*)Cout)[row * N + col] = (_Float16)c[mi][nj][r];
                else
                    ((float*)Cout)[row * N + col] = c[mi][nj][r];
            }
}

// ---------------------------------------------------------------------------
// 3-axis RoPE applied in-place to q and k slices of qkv[M_ROWS, 2304] (f16).
// One thread per rotation pair (row, head/group, axis, i).
// ---------------------------------------------------------------------------
__global__ void rope_kernel(_Float16* __restrict__ qkv) {
    size_t idx = (size_t)blockIdx.x * blockDim.x + threadIdx.x;
    const size_t total = (size_t)M_ROWS * 20 * 3 * 16;   // 15,728,640
    if (idx >= total) return;
    int i    = (int)(idx & 15);
    size_t r1 = idx >> 4;
    int axis = (int)(r1 % 3);
    size_t r2 = r1 / 3;
    int hg   = (int)(r2 % 20);          // 0..15 q heads, 16..19 kv groups
    size_t row = r2 / 20;               // 0..16383
    int s = (int)(row % S_PAT);
    int t = (int)((row / S_PAT) % T_GRID);
    int pos = (axis == 0) ? t : ((axis == 1) ? (s >> 4) : (s & 15));
    // inv_freq = 10000^(-i/16)
    float ang = (float)pos * __expf(-(float)i * 0.5756462732485114f);
    float sn, cs;
    __sincosf(ang, &sn, &cs);
    int col = ((hg < 16) ? hg * 96 : 1536 + (hg - 16) * 96) + axis * 32 + i;
    _Float16* p = qkv + row * (size_t)QKV_OUT + col;
    float x1 = (float)p[0], x2 = (float)p[16];
    p[0]  = (_Float16)(x1 * cs - x2 * sn);
    p[16] = (_Float16)(x2 * cs + x1 * sn);
}

// ---------------------------------------------------------------------------
// Flash attention per (b, t, head). 16 waves x 16 query rows = 256 queries.
// Streams 8 key-blocks of 32; scores + P*V via WMMA; online softmax with
// 16-lane shuffle reductions; P re-shaped C->A layout through per-wave LDS.
// ---------------------------------------------------------------------------
#define VSTR 40   // LDS row stride (halves) for V^T (keeps 16B alignment)
#define PSTR 40   // LDS row stride (halves) for P

__global__ __launch_bounds__(512)
void attn_kernel(const _Float16* __restrict__ qkv,
                 const unsigned char* __restrict__ mask,
                 _Float16* __restrict__ outh) {
    __shared__ __align__(16) _Float16 ldsVT[HEAD_DIM * VSTR];      // 7680 B
    __shared__ __align__(16) _Float16 ldsP[16 * 16 * PSTR];        // 20480 B

    int blk = blockIdx.x;              // ((b*T + t)*H + h)
    int h  = blk & 15;
    int bt = blk >> 4;                 // b*T + t
    int t  = bt & 31;
    int b  = bt >> 5;
    int g  = h >> 2;

    int lane = threadIdx.x & 31;
    int wave = threadIdx.x >> 5;       // 0..15
    int m = lane & 15, hi = lane >> 4;

    size_t rowBase = (size_t)bt * S_PAT;
    const _Float16* qp = qkv + (rowBase + wave * 16) * QKV_OUT + h * 96;
    const _Float16* kp = qkv + rowBase * QKV_OUT + 1536 + g * 96;
    const _Float16* vp = qkv + rowBase * QKV_OUT + 1920 + g * 96;
    const unsigned char* mp = mask + (size_t)b * N_SEQ + (size_t)t * S_PAT;

    // Preload q fragments (16 rows x 96 dims = 3 K-steps)
    v16h aq[3];
#pragma unroll
    for (int ks = 0; ks < 3; ++ks)
        aq[ks] = load_a_frag(qp + ks * 32, QKV_OUT, lane);

    v8f o[6] = {};
    float rm[8], rs[8];
#pragma unroll
    for (int r = 0; r < 8; ++r) { rm[r] = -1e30f; rs[r] = 0.0f; }

    const float scale = 0.10206207261596577f;   // 1/sqrt(96)
    _Float16* pP = ldsP + wave * 16 * PSTR;

    for (int j = 0; j < 8; ++j) {
        __syncthreads();                         // previous V^T reads done
        // Stage V^T (32 keys x 96 dims): 512 threads x 6 elements
        {
            int base = threadIdx.x * 6;
#pragma unroll
            for (int e = 0; e < 6; ++e) {
                int idx = base + e;
                int key = idx / 96;
                int dim = idx - key * 96;
                ldsVT[dim * VSTR + key] =
                    vp[(size_t)(j * 32 + key) * QKV_OUT + dim];
            }
        }
        __syncthreads();

        // Scores: q(16x96) . K^T -> two 16x16 fragments (keys j*32..+31)
        v8f s0 = {}, s1 = {};
#pragma unroll
        for (int ks = 0; ks < 3; ++ks) {
            v16h b0 = load_b_frag(kp + (size_t)(j * 32) * QKV_OUT + ks * 32,
                                  QKV_OUT, lane);
            v16h b1 = load_b_frag(kp + (size_t)(j * 32 + 16) * QKV_OUT + ks * 32,
                                  QKV_OUT, lane);
            s0 = wmma_f16(aq[ks], b0, s0);
            s1 = wmma_f16(aq[ks], b1, s1);
        }

        // Key padding mask (per column)
        float madd0 = mp[j * 32 + m]      ? 0.0f : -1e9f;
        float madd1 = mp[j * 32 + 16 + m] ? 0.0f : -1e9f;

        // Online softmax update; write P to per-wave LDS (in-order DS ops)
#pragma unroll
        for (int r = 0; r < 8; ++r) {
            float v0 = s0[r] * scale + madd0;
            float v1 = s1[r] * scale + madd1;
            float mx = fmaxf(v0, v1);
#pragma unroll
            for (int off = 1; off < 16; off <<= 1)
                mx = fmaxf(mx, __shfl_xor(mx, off, 32));
            float nm = fmaxf(rm[r], mx);
            float al = __expf(rm[r] - nm);
            rm[r] = nm;
            float p0 = __expf(v0 - nm);
            float p1 = __expf(v1 - nm);
            float ps = p0 + p1;
#pragma unroll
            for (int off = 1; off < 16; off <<= 1)
                ps += __shfl_xor(ps, off, 32);
            rs[r] = rs[r] * al + ps;
            int prow = r + 8 * hi;
            pP[prow * PSTR + m]      = (_Float16)p0;
            pP[prow * PSTR + 16 + m] = (_Float16)p1;
#pragma unroll
            for (int nd = 0; nd < 6; ++nd)
                o[nd][r] *= al;
        }

        // P(16x32) * V(32x96): one A-frag, six B-frags from staged V^T
        v16h pa = load_a_frag(pP, PSTR, lane);
#pragma unroll
        for (int nd = 0; nd < 6; ++nd) {
            v16h bv = load_b_frag(ldsVT + nd * 16 * VSTR, VSTR, lane);
            o[nd] = wmma_f16(pa, bv, o[nd]);
        }
    }

    // Normalize and store (f16) into attention-output buffer [M_ROWS, 1536]
    float inv[8];
#pragma unroll
    for (int r = 0; r < 8; ++r) inv[r] = 1.0f / rs[r];
    _Float16* op = outh + (rowBase + wave * 16) * D_MODEL + h * 96;
#pragma unroll
    for (int nd = 0; nd < 6; ++nd)
#pragma unroll
        for (int r = 0; r < 8; ++r) {
            size_t row = r + 8 * hi;
            size_t col = nd * 16 + m;
            op[row * D_MODEL + col] = (_Float16)(o[nd][r] * inv[r]);
        }
}

// ---------------------------------------------------------------------------
// Launch
// ---------------------------------------------------------------------------
extern "C" void kernel_launch(void* const* d_in, const int* in_sizes, int n_in,
                              void* d_out, int out_size, void* d_ws, size_t ws_size,
                              hipStream_t stream) {
    const float* x            = (const float*)d_in[0];
    const unsigned char* pmsk = (const unsigned char*)d_in[1];   // bool mask
    const float* w_qkv        = (const float*)d_in[2];
    const float* w_o          = (const float*)d_in[3];

    const size_t NX  = (size_t)M_ROWS * D_MODEL;    // 25,165,824
    const size_t NWQ = (size_t)QKV_OUT * D_MODEL;   //  3,538,944
    const size_t NWO = (size_t)D_MODEL * D_MODEL;   //  2,359,296

    // Workspace layout (bytes):
    //   [0)              xh  f16 [M,1536]   — reused as attention output oh
    //   [50331648)       qkvh f16 [M,2304]
    //   [125829120)      wqkvh f16 [2304,1536]
    //   [132907008)      woh  f16 [1536,1536]   (end = 137,625,600)
    char* ws = (char*)d_ws;
    _Float16* xh    = (_Float16*)(ws);
    _Float16* qkvh  = (_Float16*)(ws + 50331648ull);
    _Float16* wqkvh = (_Float16*)(ws + 125829120ull);
    _Float16* woh   = (_Float16*)(ws + 132907008ull);
    if (ws_size < 137625600ull) return;

    // 1) convert inputs to f16
    cvt_f16_kernel<<<(unsigned)((NX  + 255) / 256), 256, 0, stream>>>(x,     xh,    NX);
    cvt_f16_kernel<<<(unsigned)((NWQ + 255) / 256), 256, 0, stream>>>(w_qkv, wqkvh, NWQ);
    cvt_f16_kernel<<<(unsigned)((NWO + 255) / 256), 256, 0, stream>>>(w_o,   woh,   NWO);

    // 2) fused QKV projection: qkv = x @ w_qkv^T   [16384, 2304]
    gemm_kernel<true><<<dim3(QKV_OUT / 256, M_ROWS / 128), 256, 0, stream>>>(
        xh, wqkvh, (void*)qkvh, M_ROWS, QKV_OUT, D_MODEL);

    // 3) 3-axis RoPE on q and k, in place
    {
        const size_t total = (size_t)M_ROWS * 20 * 3 * 16;
        rope_kernel<<<(unsigned)((total + 255) / 256), 256, 0, stream>>>(qkvh);
    }

    // 4) flash attention per (b, t, head) -> oh (reuses xh buffer)
    attn_kernel<<<BATCH * T_GRID * NUM_HEADS, 512, 0, stream>>>(qkvh, pmsk, xh);

    // 5) output projection: out = oh @ w_o^T  -> f32 d_out [16384, 1536]
    gemm_kernel<false><<<dim3(D_MODEL / 256, M_ROWS / 128), 256, 0, stream>>>(
        xh, woh, d_out, M_ROWS, D_MODEL, D_MODEL);
}
